// FILM_Encoder_59425167507607
// MI455X (gfx1250) — matrix operand order repaired and checked by reading, not compiled, not run on hardware
//
#include <hip/hip_runtime.h>
#include <hip/hip_bf16.h>
#include <cstdint>
#include <cstddef>

// ---------------- problem constants (match reference) ----------------
#define N_NODESC 60000
#define N_EDGESC 960000
#define BATCHC   8192
#define HIDC     256
#define N_CLSC   64
#define K_TOPC   16
#define ETAC     0.5f
#define BN_EPSC  1e-5f
#define SIM_EPSC 1e-8f

typedef _Float16 v8h  __attribute__((ext_vector_type(8)));
typedef _Float16 v16h __attribute__((ext_vector_type(16)));
typedef float    v8f  __attribute__((ext_vector_type(8)));
typedef unsigned int u32x4 __attribute__((ext_vector_type(4)));
typedef int          i32x4 __attribute__((ext_vector_type(4)));
typedef int          i32x8 __attribute__((ext_vector_type(8)));

#if defined(__has_builtin)
#if __has_builtin(__builtin_amdgcn_tensor_load_to_lds) && \
    __has_builtin(__builtin_amdgcn_s_wait_tensorcnt)
#define HAVE_TDM 1
#endif
#endif
#ifndef HAVE_TDM
#define HAVE_TDM 0
#endif

// ---------------- small utility kernels ----------------
__global__ __launch_bounds__(256) void zero_kernel(float* __restrict__ p, size_t n) {
  size_t i = (size_t)blockIdx.x * 256 + threadIdx.x;
  if (i < n) p[i] = 0.f;
}

__global__ __launch_bounds__(256) void copy_kernel(const float* __restrict__ in,
                                                   float* __restrict__ out, size_t n) {
  size_t i = (size_t)blockIdx.x * 256 + threadIdx.x;
  if (i < n) out[i] = in[i];
}

__global__ __launch_bounds__(256) void cvt_f16_kernel(const float* __restrict__ in,
                                                      _Float16* __restrict__ out, size_t n) {
  size_t i = (size_t)blockIdx.x * 256 + threadIdx.x;
  if (i < n) out[i] = (_Float16)in[i];
}

// W (K x N, row-major f32) -> WT (N x K, row-major f16)
__global__ __launch_bounds__(256) void wtrans_kernel(const float* __restrict__ W,
                                                     _Float16* __restrict__ WT,
                                                     int K, int N) {
  long long i = (long long)blockIdx.x * 256 + threadIdx.x;
  if (i >= (long long)K * N) return;
  int k = (int)(i / N), n = (int)(i % N);
  WT[(size_t)n * K + k] = (_Float16)W[(size_t)k * N + n];
}

__global__ __launch_bounds__(256) void deg_kernel(const int* __restrict__ dst,
                                                  float* __restrict__ deg, int nE) {
  int e = blockIdx.x * 256 + threadIdx.x;
  if (e < nE) atomicAdd(&deg[dst[e]], 1.0f);
}

// ---------------- TDM helper: 2D f16 tile -> LDS (D# per ISA 8.3/8.4) ----------------
#if HAVE_TDM
__device__ __forceinline__ void tdm_load_2d_f16(
    unsigned lds_addr, const void* gaddr,
    unsigned tensor_d0, unsigned tensor_d1,
    unsigned tile_d0, unsigned tile_d1, unsigned stride0) {
  unsigned long long ga = (unsigned long long)gaddr;
  u32x4 g0;
  g0.x = 1u;                                    // count=1 (valid user descriptor)
  g0.y = lds_addr;                              // LDS byte address
  g0.z = (unsigned)(ga & 0xffffffffu);          // global_addr[31:0]
  g0.w = (unsigned)((ga >> 32) & 0x01ffffffu)   // global_addr[56:32]
         | (2u << 30);                          // type=2 ("image")
  i32x8 g1;
  // data_size=1 (2 bytes); pad_enable, pad_interval=6 (128 dwords = one 256-elem
  // f16 row), pad_amount=3 (4 dwords = 8 halves) -> LDS row stride 264 halves.
  g1[0] = (int)((1u << 16) | (1u << 20) | (6u << 22) | (3u << 25));
  g1[1] = (int)((tensor_d0 & 0xffffu) << 16);                       // dim0[15:0]
  g1[2] = (int)(((tensor_d0 >> 16) & 0xffffu) |
                ((tensor_d1 & 0xffffu) << 16));                     // dim0[31:16] | dim1[15:0]
  g1[3] = (int)(((tensor_d1 >> 16) & 0xffffu) |
                ((tile_d0 & 0xffffu) << 16));                       // dim1[31:16] | tile0
  g1[4] = (int)(tile_d1 & 0xffffu);                                 // tile1 | tile2=0
  g1[5] = (int)stride0;                                             // dim0_stride[31:0]
  g1[6] = 0;                                                        // stride0[47:32]|stride1 lo
  g1[7] = 0;
  i32x4 z4 = {0, 0, 0, 0};
#if defined(__clang_major__) && (__clang_major__ >= 23)
  i32x8 z8 = {0, 0, 0, 0, 0, 0, 0, 0};
  __builtin_amdgcn_tensor_load_to_lds(g0, g1, z4, z4, z8, 0);
#else
  __builtin_amdgcn_tensor_load_to_lds(g0, g1, z4, z4, 0);
#endif
}
#endif

// ---------------- WMMA GEMM with LDS staging ----------------
// C(MxN,f32) = A(MxK,f16 row-major) * BT(NxK,f16 row-major)^T, K == 256.
// Workgroup: 8 waves = 2x4 wave grid; block tile 64(M) x 128(N); each wave owns a
// 32x32 tile = 2x2 v_wmma_f32_16x16x32_f16 tiles. A/B panels staged to LDS by the
// Tensor Data Mover (pad_amount gives a 528B row pitch -> bank-conflict-free
// ds_load_b128 fragment reads across lanes).
#define LDS_PITCH 264   // halves per row: 256 data + 8 pad
#define KDIM 256

__device__ __forceinline__ v16h frag_a(const _Float16* p, int sel) {
  // A fragment (ISA 7.12.2): halves 0..7 <- K = base + sel*8, halves 8..15 <- +16
  v8h lo = *(const v8h*)(p + sel * 8);
  v8h hi = *(const v8h*)(p + 16 + sel * 8);
  return __builtin_shufflevector(lo, hi, 0,1,2,3,4,5,6,7,8,9,10,11,12,13,14,15);
}
__device__ __forceinline__ v16h frag_b(const _Float16* p, int sel) {
  // B fragment: halves 0..7 <- K = base + sel*8, halves 8..15 <- +16
  v8h lo = *(const v8h*)(p + sel * 8);
  v8h hi = *(const v8h*)(p + 16 + sel * 8);
  return __builtin_shufflevector(lo, hi, 0,1,2,3,4,5,6,7,8,9,10,11,12,13,14,15);
}

__global__ __launch_bounds__(256) void gemm_f16_wmma(
    const _Float16* __restrict__ A,   // M x 256
    const _Float16* __restrict__ BT,  // N x 256
    float* __restrict__ C,            // M x N (ldc)
    int M, int N, int ldc) {
  __shared__ _Float16 ash[64 * LDS_PITCH];
  __shared__ _Float16 bsh[128 * LDS_PITCH];
  const int lane = threadIdx.x & 31;
  const int wave = threadIdx.x >> 5;
  const int row0 = blockIdx.x * 64;
  const int col0 = blockIdx.y * 128;

#if HAVE_TDM
  if (threadIdx.x < 32) {   // one wave drives both DMA descriptors
    tdm_load_2d_f16((unsigned)(size_t)(void*)ash, A + (size_t)row0 * KDIM,
                    KDIM, (unsigned)(M - row0), KDIM, 64, KDIM);
    tdm_load_2d_f16((unsigned)(size_t)(void*)bsh, BT + (size_t)col0 * KDIM,
                    KDIM, (unsigned)(N - col0), KDIM, 128, KDIM);
    __builtin_amdgcn_s_wait_tensorcnt(0);
  }
#else
  // fallback staging: cooperative vector copies with zero padding
  for (int idx = threadIdx.x; idx < 64 * 32; idx += 256) {
    int r = idx >> 5, ch = idx & 31;
    v8h v = {};
    if (row0 + r < M) v = *(const v8h*)(A + (size_t)(row0 + r) * KDIM + ch * 8);
    *(v8h*)(ash + r * LDS_PITCH + ch * 8) = v;
  }
  for (int idx = threadIdx.x; idx < 128 * 32; idx += 256) {
    int r = idx >> 5, ch = idx & 31;
    v8h v = {};
    if (col0 + r < N) v = *(const v8h*)(BT + (size_t)(col0 + r) * KDIM + ch * 8);
    *(v8h*)(bsh + r * LDS_PITCH + ch * 8) = v;
  }
#endif
  __syncthreads();

  const int wr = wave >> 2;          // 0..1 -> 32-row slab
  const int wc = wave & 3;           // 0..3 -> 32-col slab
  const int ml = lane & 15;
  const int sel = lane >> 4;
  const _Float16* a0p = ash + (wr * 32 + ml) * LDS_PITCH;
  const _Float16* a1p = ash + (wr * 32 + 16 + ml) * LDS_PITCH;
  const _Float16* b0p = bsh + (wc * 32 + ml) * LDS_PITCH;
  const _Float16* b1p = bsh + (wc * 32 + 16 + ml) * LDS_PITCH;

  v8f acc00 = {}, acc01 = {}, acc10 = {}, acc11 = {};
#pragma unroll
  for (int kk = 0; kk < KDIM; kk += 32) {
    v16h a0 = frag_a(a0p + kk, sel);
    v16h a1 = frag_a(a1p + kk, sel);
    v16h b0 = frag_b(b0p + kk, sel);
    v16h b1 = frag_b(b1p + kk, sel);
    acc00 = __builtin_amdgcn_wmma_f32_16x16x32_f16(false, a0, false, b0, (short)0, acc00, false, false);
    acc01 = __builtin_amdgcn_wmma_f32_16x16x32_f16(false, a0, false, b1, (short)0, acc01, false, false);
    acc10 = __builtin_amdgcn_wmma_f32_16x16x32_f16(false, a1, false, b0, (short)0, acc10, false, false);
    acc11 = __builtin_amdgcn_wmma_f32_16x16x32_f16(false, a1, false, b1, (short)0, acc11, false, false);
  }

  // D layout: VGPR r -> row m = r + 8*sel, col n = lane&15
  const v8f accs[2][2] = {{acc00, acc01}, {acc10, acc11}};
#pragma unroll
  for (int tm = 0; tm < 2; ++tm) {
#pragma unroll
    for (int tn = 0; tn < 2; ++tn) {
      int col = col0 + wc * 32 + tn * 16 + ml;
      int rbase = row0 + wr * 32 + tm * 16 + sel * 8;
      if (col < N) {
        float* cp = C + (size_t)rbase * ldc + col;
#pragma unroll
        for (int r = 0; r < 8; ++r)
          if (rbase + r < M) cp[(size_t)r * ldc] = accs[tm][tn][r];
      }
    }
  }
}

// ---------------- edge message + scatter-add ----------------
template <int F, bool RELU>
__global__ __launch_bounds__(256) void edge_agg_kernel(
    const int* __restrict__ src, const int* __restrict__ dst,
    const float* __restrict__ T, const float* __restrict__ BG2,
    float* __restrict__ AGG, int nE) {
  constexpr int LPE = F / 4;
  long long t = (long long)blockIdx.x * 256 + threadIdx.x;
  int e = (int)(t / LPE);
  if (e >= nE) return;
  int c = (int)(t % LPE) * 4;
  int s = src[e], d = dst[e];
  const float4 tv = *(const float4*)(T + (size_t)s * F + c);
  const float4 bv = *(const float4*)(BG2 + (size_t)d * (2 * F) + c);       // beta
  const float4 gv = *(const float4*)(BG2 + (size_t)d * (2 * F) + F + c);   // gamma
  float m0 = gv.x * tv.x + bv.x;
  float m1 = gv.y * tv.y + bv.y;
  float m2 = gv.z * tv.z + bv.z;
  float m3 = gv.w * tv.w + bv.w;
  if (RELU) {
    m0 = fmaxf(m0, 0.f); m1 = fmaxf(m1, 0.f);
    m2 = fmaxf(m2, 0.f); m3 = fmaxf(m3, 0.f);
  }
  float* a = AGG + (size_t)d * F + c;
  atomicAdd(a + 0, m0);
  atomicAdd(a + 1, m1);
  atomicAdd(a + 2, m2);
  atomicAdd(a + 3, m3);
}

// ---------------- self path + aggregation + BN ----------------
__global__ __launch_bounds__(256) void combine_bn_kernel(
    const float* __restrict__ S, const float* __restrict__ BG,
    const float* __restrict__ AGG, const float* __restrict__ deg,
    const float* __restrict__ g, const float* __restrict__ b,
    const float* __restrict__ mu, const float* __restrict__ var,
    float* __restrict__ Hn) {
  size_t i = (size_t)blockIdx.x * 256 + threadIdx.x;
  if (i >= (size_t)N_NODESC * HIDC) return;
  int n = (int)(i / HIDC), c = (int)(i % HIDC);
  float self = BG[(size_t)n * (2 * HIDC) + HIDC + c] * S[i] +
               BG[(size_t)n * (2 * HIDC) + c];
  self = fmaxf(self, 0.f);
  float v = self + AGG[i] / fmaxf(deg[n], 1.f);
  Hn[i] = (v - mu[c]) * rsqrtf(var[c] + BN_EPSC) * g[c] + b[c];
}

__global__ __launch_bounds__(256) void combine_final_kernel(
    const float* __restrict__ S, const float* __restrict__ BG,
    const float* __restrict__ AGG, const float* __restrict__ deg,
    float* __restrict__ lc) {
  size_t i = (size_t)blockIdx.x * 256 + threadIdx.x;
  if (i >= (size_t)BATCHC * N_CLSC) return;
  int n = (int)(i / N_CLSC), c = (int)(i % N_CLSC);
  float self = BG[(size_t)n * (2 * N_CLSC) + N_CLSC + c] * S[(size_t)n * N_CLSC + c] +
               BG[(size_t)n * (2 * N_CLSC) + c];
  lc[i] = self + AGG[(size_t)n * N_CLSC + c] / fmaxf(deg[n], 1.f);
}

// in-place row log_softmax over 64 classes; one wave32 per row, 2 cls/lane
__global__ __launch_bounds__(256) void logsoftmax64_kernel(float* __restrict__ buf) {
  int wave = threadIdx.x >> 5, lane = threadIdx.x & 31;
  int row = blockIdx.x * 8 + wave;
  if (row >= BATCHC) return;
  float* r = buf + (size_t)row * N_CLSC;
  float a = r[lane], b = r[lane + 32];
  float m = fmaxf(a, b);
#pragma unroll
  for (int off = 16; off; off >>= 1) m = fmaxf(m, __shfl_xor(m, off, 32));
  float s = expf(a - m) + expf(b - m);
#pragma unroll
  for (int off = 16; off; off >>= 1) s += __shfl_xor(s, off, 32);
  float ls = logf(s);
  r[lane] = a - m - ls;
  r[lane + 32] = b - m - ls;
}

// L2-normalize first BATCH embedding rows into f16; one wave per row
__global__ __launch_bounds__(256) void ennorm_kernel(const float* __restrict__ emb,
                                                     _Float16* __restrict__ en) {
  int wave = threadIdx.x >> 5, lane = threadIdx.x & 31;
  int row = blockIdx.x * 8 + wave;
  if (row >= BATCHC) return;
  const float* e = emb + (size_t)row * HIDC;
  float vals[8];
  float s = 0.f;
#pragma unroll
  for (int j = 0; j < 8; ++j) { vals[j] = e[lane + 32 * j]; s += vals[j] * vals[j]; }
#pragma unroll
  for (int off = 16; off; off >>= 1) s += __shfl_xor(s, off, 32);
  float inv = 1.f / fmaxf(sqrtf(s), SIM_EPSC);
  _Float16* o = en + (size_t)row * HIDC;
#pragma unroll
  for (int j = 0; j < 8; ++j) o[lane + 32 * j] = (_Float16)(vals[j] * inv);
}

// ---------------- fused top-k + label fusion + log_softmax + blend ----------------
__device__ __forceinline__ void topk_insert(float* v, int* id, float nv, int ni) {
  if (nv <= v[K_TOPC - 1]) return;
  int k = K_TOPC - 1;
  while (k > 0 && v[k - 1] < nv) { v[k] = v[k - 1]; id[k] = id[k - 1]; --k; }
  v[k] = nv; id[k] = ni;
}

__global__ __launch_bounds__(256) void topk_fused_kernel(
    const float* __restrict__ sim,    // stripe: gridDim.x rows x BATCH cols
    const int* __restrict__ y,
    const float* __restrict__ p_lc,   // BATCH x 64 (already log-softmaxed)
    float* __restrict__ out_final,    // BATCH x 64
    int rowBase) {
  __shared__ float sv[256 * K_TOPC];
  __shared__ int   si[256 * K_TOPC];
  __shared__ float sv2[32 * K_TOPC];
  __shared__ int   si2[32 * K_TOPC];
  __shared__ float fused[N_CLSC];
  const int t = threadIdx.x;
  const int row = rowBase + blockIdx.x;
  const float* srow = sim + (size_t)blockIdx.x * BATCHC;

  float v[K_TOPC]; int id[K_TOPC];
#pragma unroll
  for (int k = 0; k < K_TOPC; ++k) { v[k] = -3.0e38f; id[k] = 0; }
  for (int c = t; c < BATCHC; c += 256) topk_insert(v, id, srow[c], c);
#pragma unroll
  for (int k = 0; k < K_TOPC; ++k) { sv[t * K_TOPC + k] = v[k]; si[t * K_TOPC + k] = id[k]; }
  if (t < N_CLSC) fused[t] = 0.f;
  __syncthreads();

  if (t < 32) {
    float mv[K_TOPC]; int mi[K_TOPC];
#pragma unroll
    for (int k = 0; k < K_TOPC; ++k) { mv[k] = -3.0e38f; mi[k] = 0; }
    for (int j = t * 8; j < t * 8 + 8; ++j)
      for (int k = 0; k < K_TOPC; ++k)
        topk_insert(mv, mi, sv[j * K_TOPC + k], si[j * K_TOPC + k]);
#pragma unroll
    for (int k = 0; k < K_TOPC; ++k) { sv2[t * K_TOPC + k] = mv[k]; si2[t * K_TOPC + k] = mi[k]; }
  }
  __syncthreads();

  if (t == 0) {
    float mv[K_TOPC]; int mi[K_TOPC];
#pragma unroll
    for (int k = 0; k < K_TOPC; ++k) { mv[k] = -3.0e38f; mi[k] = 0; }
    for (int j = 0; j < 32; ++j)
      for (int k = 0; k < K_TOPC; ++k)
        topk_insert(mv, mi, sv2[j * K_TOPC + k], si2[j * K_TOPC + k]);
    for (int k = 0; k < K_TOPC; ++k) fused[y[mi[k]]] += expf(mv[k]);
  }
  __syncthreads();

  if (t < 32) {
    float a = fused[t], b = fused[t + 32];
    float m = fmaxf(a, b);
#pragma unroll
    for (int off = 16; off; off >>= 1) m = fmaxf(m, __shfl_xor(m, off, 32));
    float s = expf(a - m) + expf(b - m);
#pragma unroll
    for (int off = 16; off; off >>= 1) s += __shfl_xor(s, off, 32);
    float ls = logf(s);
    const float* pl = p_lc + (size_t)row * N_CLSC;
    float* of = out_final + (size_t)row * N_CLSC;
    of[t]      = ETAC * pl[t]      + (1.f - ETAC) * (a - m - ls);
    of[t + 32] = ETAC * pl[t + 32] + (1.f - ETAC) * (b - m - ls);
  }
}

// ---------------- host orchestration ----------------
static inline size_t alignUp256(size_t v) { return (v + 255) & ~(size_t)255; }
static inline dim3 blk1(long long n) { return dim3((unsigned)((n + 255) / 256)); }
static inline dim3 gemmGrid(int M, int N) {
  return dim3((unsigned)((M + 63) / 64), (unsigned)((N + 127) / 128));
}

extern "C" void kernel_launch(void* const* d_in, const int* in_sizes, int n_in,
                              void* d_out, int out_size, void* d_ws, size_t ws_size,
                              hipStream_t stream) {
  (void)in_sizes; (void)n_in; (void)out_size; (void)ws_size;
  const float* x    = (const float*)d_in[0];
  const int*   ei   = (const int*)d_in[1];
  const int*   y    = (const int*)d_in[2];
  // d_in[3] = batch_size (compile-time constant BATCHC)
  const float* Wls  = (const float*)d_in[4];
  const float* Wfs  = (const float*)d_in[5];
  const float* Wl   = (const float*)d_in[6];
  const float* Wf   = (const float*)d_in[7];
  const float* WlsL = (const float*)d_in[8];
  const float* WfsL = (const float*)d_in[9];
  const float* WlL  = (const float*)d_in[10];
  const float* WfL  = (const float*)d_in[11];
  const float* bng  = (const float*)d_in[12];
  const float* bnb  = (const float*)d_in[13];
  const float* bnm  = (const float*)d_in[14];
  const float* bnv  = (const float*)d_in[15];
  const int* srcIdx = ei;
  const int* dstIdx = ei + N_EDGESC;

  float* out_final = (float*)d_out;                               // BATCH x 64
  float* out_emb   = (float*)d_out + (size_t)BATCHC * N_CLSC;     // BATCH x 256

  // deterministic bump allocator over d_ws (~530 MB required)
  char* wp = (char*)d_ws;
  auto take = [&](size_t bytes) { char* r = wp; wp += alignUp256(bytes); return r; };
  float*    deg = (float*)take((size_t)N_NODESC * 4);
  _Float16* h16 = (_Float16*)take((size_t)N_NODESC * HIDC * 2);
  float*    Hn  = (float*)take((size_t)N_NODESC * HIDC * 4);
  float*    S   = (float*)take((size_t)N_NODESC * HIDC * 4);
  float*    BG  = (float*)take((size_t)N_NODESC * 2 * HIDC * 4);
  float*    T   = (float*)take((size_t)N_NODESC * HIDC * 4);
  float*    BG2 = (float*)take((size_t)N_NODESC * 2 * HIDC * 4);
  float*    AGG = (float*)take((size_t)N_NODESC * HIDC * 4);
  float*    lc  = (float*)take((size_t)BATCHC * N_CLSC * 4);
  _Float16 *WlsT[3], *WfsT[3], *WlT[3], *WfT[3];
  for (int l = 0; l < 3; ++l) {
    WlsT[l] = (_Float16*)take((size_t)HIDC * HIDC * 2);
    WfsT[l] = (_Float16*)take((size_t)HIDC * 2 * HIDC * 2);
    WlT[l]  = (_Float16*)take((size_t)HIDC * HIDC * 2);
    WfT[l]  = (_Float16*)take((size_t)HIDC * 2 * HIDC * 2);
  }
  _Float16* WlsLT = (_Float16*)take((size_t)HIDC * N_CLSC * 2);
  _Float16* WfsLT = (_Float16*)take((size_t)HIDC * 2 * N_CLSC * 2);
  _Float16* WlLT  = (_Float16*)take((size_t)HIDC * N_CLSC * 2);
  _Float16* WfLT  = (_Float16*)take((size_t)HIDC * 2 * N_CLSC * 2);
  _Float16* en16   = (_Float16*)T;   // reused after final combine
  float*    simbuf = BG;             // reused: 2048x8192 f32 = 67 MB <= 123 MB

  // 1) degrees
  zero_kernel<<<blk1(N_NODESC), 256, 0, stream>>>(deg, (size_t)N_NODESC);
  deg_kernel<<<blk1(N_EDGESC), 256, 0, stream>>>(dstIdx, deg, N_EDGESC);

  // 2) transpose + f16-quantize all weights once
  for (int l = 0; l < 3; ++l) {
    wtrans_kernel<<<blk1((long long)HIDC * HIDC), 256, 0, stream>>>(
        Wls + (size_t)l * HIDC * HIDC, WlsT[l], HIDC, HIDC);
    wtrans_kernel<<<blk1((long long)HIDC * 2 * HIDC), 256, 0, stream>>>(
        Wfs + (size_t)l * HIDC * 2 * HIDC, WfsT[l], HIDC, 2 * HIDC);
    wtrans_kernel<<<blk1((long long)HIDC * HIDC), 256, 0, stream>>>(
        Wl + (size_t)l * HIDC * HIDC, WlT[l], HIDC, HIDC);
    wtrans_kernel<<<blk1((long long)HIDC * 2 * HIDC), 256, 0, stream>>>(
        Wf + (size_t)l * HIDC * 2 * HIDC, WfT[l], HIDC, 2 * HIDC);
  }
  wtrans_kernel<<<blk1((long long)HIDC * N_CLSC), 256, 0, stream>>>(WlsL, WlsLT, HIDC, N_CLSC);
  wtrans_kernel<<<blk1((long long)HIDC * 2 * N_CLSC), 256, 0, stream>>>(WfsL, WfsLT, HIDC, 2 * N_CLSC);
  wtrans_kernel<<<blk1((long long)HIDC * N_CLSC), 256, 0, stream>>>(WlL, WlLT, HIDC, N_CLSC);
  wtrans_kernel<<<blk1((long long)HIDC * 2 * N_CLSC), 256, 0, stream>>>(WfL, WfLT, HIDC, 2 * N_CLSC);

  // 3) h0 = x -> f16
  cvt_f16_kernel<<<blk1((long long)N_NODESC * HIDC), 256, 0, stream>>>(
      x, h16, (size_t)N_NODESC * HIDC);

  // 4) three FiLM layers + BN
  for (int l = 0; l < 3; ++l) {
    gemm_f16_wmma<<<gemmGrid(N_NODESC, HIDC), 256, 0, stream>>>(
        h16, WlsT[l], S, N_NODESC, HIDC, HIDC);
    gemm_f16_wmma<<<gemmGrid(N_NODESC, 2 * HIDC), 256, 0, stream>>>(
        h16, WfsT[l], BG, N_NODESC, 2 * HIDC, 2 * HIDC);
    gemm_f16_wmma<<<gemmGrid(N_NODESC, HIDC), 256, 0, stream>>>(
        h16, WlT[l], T, N_NODESC, HIDC, HIDC);
    gemm_f16_wmma<<<gemmGrid(N_NODESC, 2 * HIDC), 256, 0, stream>>>(
        h16, WfT[l], BG2, N_NODESC, 2 * HIDC, 2 * HIDC);
    zero_kernel<<<blk1((long long)N_NODESC * HIDC), 256, 0, stream>>>(
        AGG, (size_t)N_NODESC * HIDC);
    edge_agg_kernel<HIDC, true><<<blk1((long long)N_EDGESC * (HIDC / 4)), 256, 0, stream>>>(
        srcIdx, dstIdx, T, BG2, AGG, N_EDGESC);
    combine_bn_kernel<<<blk1((long long)N_NODESC * HIDC), 256, 0, stream>>>(
        S, BG, AGG, deg, bng + (size_t)l * HIDC, bnb + (size_t)l * HIDC,
        bnm + (size_t)l * HIDC, bnv + (size_t)l * HIDC, Hn);
    cvt_f16_kernel<<<blk1((long long)N_NODESC * HIDC), 256, 0, stream>>>(
        Hn, h16, (size_t)N_NODESC * HIDC);
  }

  // 5) final FiLM layer -> lc (first BATCH rows)
  gemm_f16_wmma<<<gemmGrid(N_NODESC, N_CLSC), 256, 0, stream>>>(
      h16, WlsLT, S, N_NODESC, N_CLSC, N_CLSC);
  gemm_f16_wmma<<<gemmGrid(N_NODESC, 2 * N_CLSC), 256, 0, stream>>>(
      h16, WfsLT, BG, N_NODESC, 2 * N_CLSC, 2 * N_CLSC);
  gemm_f16_wmma<<<gemmGrid(N_NODESC, N_CLSC), 256, 0, stream>>>(
      h16, WlLT, T, N_NODESC, N_CLSC, N_CLSC);
  gemm_f16_wmma<<<gemmGrid(N_NODESC, 2 * N_CLSC), 256, 0, stream>>>(
      h16, WfLT, BG2, N_NODESC, 2 * N_CLSC, 2 * N_CLSC);
  zero_kernel<<<blk1((long long)N_NODESC * N_CLSC), 256, 0, stream>>>(
      AGG, (size_t)N_NODESC * N_CLSC);
  edge_agg_kernel<N_CLSC, false><<<blk1((long long)N_EDGESC * (N_CLSC / 4)), 256, 0, stream>>>(
      srcIdx, dstIdx, T, BG2, AGG, N_EDGESC);
  combine_final_kernel<<<blk1((long long)BATCHC * N_CLSC), 256, 0, stream>>>(
      S, BG, AGG, deg, lc);
  logsoftmax64_kernel<<<BATCHC / 8, 256, 0, stream>>>(lc);   // lc -> p_lc in place

  // 6) embedding output + normalized f16 embeddings
  copy_kernel<<<blk1((long long)BATCHC * HIDC), 256, 0, stream>>>(
      Hn, out_emb, (size_t)BATCHC * HIDC);
  ennorm_kernel<<<BATCHC / 8, 256, 0, stream>>>(Hn, en16);

  // 7) sim = en @ en^T in 2048-row stripes (64 MB stripe lives in 192 MB L2),
  //    fused top-16 + exp-weighted label fusion + log_softmax + blend
  for (int sblk = 0; sblk < 4; ++sblk) {
    gemm_f16_wmma<<<gemmGrid(2048, BATCHC), 256, 0, stream>>>(
        en16 + (size_t)sblk * 2048 * HIDC, en16, simbuf, 2048, BATCHC, BATCHC);
    topk_fused_kernel<<<2048, 256, 0, stream>>>(simbuf, y, lc, out_final, sblk * 2048);
  }
}
